// MyParticleNetwork_31988916420944
// MI455X (gfx1250) — matrix operand order repaired and checked by reading, not compiled, not run on hardware
//
#include <hip/hip_runtime.h>
#include <hip/hip_bf16.h>
#include <math.h>

#define N_PTS 4096
#define K_NBR 80
#define CH    128
#define DM    128
#define NCELL 64
#define FFN   512
#define TDM_CHUNK 512

typedef __bf16 bf16;
typedef __attribute__((ext_vector_type(16))) __bf16 v16bf;
typedef __attribute__((ext_vector_type(8)))  __bf16 v8bf;
typedef __attribute__((ext_vector_type(8)))  float  v8f;
typedef __attribute__((ext_vector_type(4)))  unsigned u32x4;
typedef __attribute__((ext_vector_type(8)))  unsigned u32x8;

// ---------------------------------------------------------------------------
// small helpers
// ---------------------------------------------------------------------------
__device__ __forceinline__ float red16_sum(float v) {
  v += __shfl_xor(v, 1); v += __shfl_xor(v, 2);
  v += __shfl_xor(v, 4); v += __shfl_xor(v, 8);
  return v;
}
__device__ __forceinline__ float red16_max(float v) {
  v = fmaxf(v, __shfl_xor(v, 1)); v = fmaxf(v, __shfl_xor(v, 2));
  v = fmaxf(v, __shfl_xor(v, 4)); v = fmaxf(v, __shfl_xor(v, 8));
  return v;
}
__device__ __forceinline__ float red32_sum(float v) {
  v += __shfl_xor(v, 1); v += __shfl_xor(v, 2);
  v += __shfl_xor(v, 4); v += __shfl_xor(v, 8);
  v += __shfl_xor(v, 16);
  return v;
}

// ---------------------------------------------------------------------------
// weight conversion
// ---------------------------------------------------------------------------
__global__ void convert_f32_bf16(const float* __restrict__ src,
                                 bf16* __restrict__ dst, int n) {
  int e = blockIdx.x * 256 + threadIdx.x;
  if (e < n) dst[e] = (bf16)src[e];
}

// Wf (64,128,128)[g][c][o] -> WfT[o][g*128+c]  (BT layout for the GEMM)
__global__ void convert_Wf_T(const float* __restrict__ W,
                             bf16* __restrict__ WT) {
  int e = blockIdx.x * 256 + threadIdx.x;          // e = o*8192 + gc
  if (e >= NCELL * CH * DM) return;
  int o  = e >> 13;
  int gc = e & 8191;
  WT[e] = (bf16)W[(size_t)gc * DM + o];
}

// ---------------------------------------------------------------------------
// cconv stage 1: per point, build A[n] (64 x 128) in LDS, emit bf16
// ---------------------------------------------------------------------------
__global__ void __launch_bounds__(256)
cconv_gather(const float* __restrict__ feat, const float* __restrict__ pos,
             const int* __restrict__ idx, const unsigned char* __restrict__ mask,
             bf16* __restrict__ Aout) {
  __shared__ float Asm[NCELL * CH];                // 32 KB
  const int n = blockIdx.x;
  const int tid = threadIdx.x;
  for (int i = tid; i < NCELL * CH; i += 256) Asm[i] = 0.f;
  __syncthreads();

  const float INV_R = 1.0f / ((1.5f * 6.0f * 0.025f) * 0.5f);
  const float cx = pos[n * 3 + 0], cy = pos[n * 3 + 1], cz = pos[n * 3 + 2];
  const int wave = tid >> 5, lane = tid & 31;

  for (int k = wave; k < K_NBR; k += 8) {
    int j = idx[n * K_NBR + k];
    float m = mask[n * K_NBR + k] ? 1.f : 0.f;
    float rx = (pos[j * 3 + 0] - cx) * INV_R;
    float ry = (pos[j * 3 + 1] - cy) * INV_R;
    float rz = (pos[j * 3 + 2] - cz) * INV_R;
    float r2 = rx * rx + ry * ry + rz * rz;
    float w1 = 1.f - r2;
    float win = fminf(fmaxf(w1 * w1 * w1, 0.f), 1.f) * m;
    float nrm = sqrtf(fmaxf(r2, 1e-12f));
    float linf = fmaxf(fmaxf(fabsf(rx), fabsf(ry)), fmaxf(fabsf(rz), 1e-9f));
    float s = nrm / linf;
    float gx = fminf(fmaxf((rx * s + 1.f) * 1.5f, 0.f), 3.f);
    float gy = fminf(fmaxf((ry * s + 1.f) * 1.5f, 0.f), 3.f);
    float gz = fminf(fmaxf((rz * s + 1.f) * 1.5f, 0.f), 3.f);
    float f0x = fminf(floorf(gx), 2.f);
    float f0y = fminf(floorf(gy), 2.f);
    float f0z = fminf(floorf(gz), 2.f);
    float tx = gx - f0x, ty = gy - f0y, tz = gz - f0z;
    int ix = (int)f0x, iy = (int)f0y, iz = (int)f0z;
    const float* frow = feat + (size_t)j * CH;
    #pragma unroll
    for (int d = 0; d < 8; ++d) {
      int dx = (d >> 2) & 1, dy = (d >> 1) & 1, dz = d & 1;
      float cw = (dx ? tx : 1.f - tx) * (dy ? ty : 1.f - ty) * (dz ? tz : 1.f - tz);
      int cell = (ix + dx) * 16 + (iy + dy) * 4 + (iz + dz);
      float w = win * cw;
      for (int c = lane; c < CH; c += 32)
        atomicAdd(&Asm[cell * CH + c], w * frow[c]);
    }
  }
  __syncthreads();
  bf16* out = Aout + (size_t)n * (NCELL * CH);
  for (int i = tid; i < NCELL * CH; i += 256) out[i] = (bf16)Asm[i];
}

// ---------------------------------------------------------------------------
// generic WMMA GEMM:  out[M][Nn] = A[M][Kd] (bf16) * BT[Nn][Kd]^T + bias
// block = 128 threads (4 waves), wave -> one 16x16 C tile
// transB: bf16 output written transposed [Nn][M] (for attention V)
// ---------------------------------------------------------------------------
__global__ void __launch_bounds__(128)
gemm_bf16(const bf16* __restrict__ A, const bf16* __restrict__ BT,
          const float* __restrict__ bias, const float* __restrict__ resid,
          float* __restrict__ outF, bf16* __restrict__ outB,
          int M, int Kd, int Nn, int relu, int transB) {
  const int lane = threadIdx.x & 31;
  const int wave = threadIdx.x >> 5;
  const int mt = blockIdx.x * 4 + wave;
  const int nt = blockIdx.y;
  const int half = lane >> 4, l16 = lane & 15;

  const bf16* arow = A  + (size_t)(mt * 16 + l16) * Kd + half * 16;
  const bf16* brow = BT + (size_t)(nt * 16 + l16) * Kd + half * 16;

  v8f acc = {};
  for (int kc = 0; kc < Kd; kc += 32) {
    __builtin_prefetch(arow + kc + 256, 0, 0);
    v8bf alo = *(const v8bf*)(arow + kc);
    v8bf ahi = *(const v8bf*)(arow + kc + 8);
    v16bf b  = *(const v16bf*)(brow + kc);
    v16bf a;
    #pragma unroll
    for (int j = 0; j < 8; ++j) { a[j] = alo[j]; a[8 + j] = ahi[j]; }
    acc = __builtin_amdgcn_wmma_f32_16x16x32_bf16(false, a, false, b,
                                                  (short)0, acc, false, false);
  }
  #pragma unroll
  for (int r = 0; r < 8; ++r) {
    int m  = mt * 16 + r + half * 8;
    int nn = nt * 16 + l16;
    float v = acc[r];
    if (bias)  v += bias[nn];
    if (resid) v += resid[(size_t)m * Nn + nn];
    if (relu)  v = fmaxf(v, 0.f);
    if (outF) outF[(size_t)m * Nn + nn] = v;
    if (outB) {
      if (transB) outB[(size_t)nn * M + m] = (bf16)v;
      else        outB[(size_t)m * Nn + nn] = (bf16)v;
    }
  }
}

// ---------------------------------------------------------------------------
// cconv GEMM with TDM (tensor_load_to_lds) staging of the shared B tile.
// Fixed shape: M=4096, Kd=8192, Nn=128. Wave 0 DMAs a 16x512 bf16 tile of
// BT into LDS (D# per ISA 08_async_tensor 8.3/8.4), waits TENSORcnt, then
// all 4 waves consume B-fragments from LDS.
// ---------------------------------------------------------------------------
__global__ void __launch_bounds__(128)
gemm_cconv_tdm(const bf16* __restrict__ A, const bf16* __restrict__ BT,
               const float* __restrict__ bias, float* __restrict__ outF) {
  __shared__ __align__(32) bf16 bsm[16 * TDM_CHUNK];   // 16 KB
  const int Kd = NCELL * CH;                           // 8192
  const int Nn = DM;
  const int lane = threadIdx.x & 31;
  const int wave = threadIdx.x >> 5;
  const int mt = blockIdx.x * 4 + wave;
  const int nt = blockIdx.y;
  const int half = lane >> 4, l16 = lane & 15;

  const bf16* arow = A + (size_t)(mt * 16 + l16) * Kd + half * 16;
  const unsigned lds_base =
      (unsigned)(size_t)(__attribute__((address_space(3))) const void*)bsm;

  v8f acc = {};
  for (int chunk = 0; chunk < Kd; chunk += TDM_CHUNK) {
    if (wave == 0) {
      unsigned long long ga =
          (unsigned long long)(const void*)(BT + (size_t)(nt * 16) * Kd + chunk);
      u32x4 g0;
      g0[0] = 1u;                                   // count=1 valid descriptor
      g0[1] = lds_base;                             // lds_addr (bytes)
      g0[2] = (unsigned)(ga & 0xffffffffull);       // global_addr lo
      g0[3] = (unsigned)((ga >> 32) & 0x1ffffffull) // global_addr hi (57b)
              | 0x80000000u;                        // type=2 ("image")
      u32x8 g1;
      g1[0] = 1u << 16;                             // data_size=1 -> 2 bytes
      g1[1] = ((unsigned)Kd & 0xffffu) << 16;       // tensor_dim0 lo16
      g1[2] = ((unsigned)Kd >> 16) | (16u << 16);   // tensor_dim0 hi / dim1 lo
      g1[3] = ((unsigned)TDM_CHUNK) << 16;          // dim1 hi=0, tile_dim0=512
      g1[4] = 16u;                                  // tile_dim1=16, tile_dim2=0
      g1[5] = (unsigned)Kd;                         // tensor_dim0_stride lo32
      g1[6] = 0u;                                   // stride0 hi16, stride1 lo16
      g1[7] = 0u;                                   // stride1 hi32
      asm volatile("tensor_load_to_lds %0, %1" :: "s"(g0), "s"(g1) : "memory");
      __builtin_amdgcn_s_wait_tensorcnt(0);
    }
    __syncthreads();
    #pragma unroll 4
    for (int kc = 0; kc < TDM_CHUNK; kc += 32) {
      __builtin_prefetch(arow + chunk + kc + 256, 0, 0);
      v8bf alo = *(const v8bf*)(arow + chunk + kc);
      v8bf ahi = *(const v8bf*)(arow + chunk + kc + 8);
      const v16bf b = *(const v16bf*)&bsm[l16 * TDM_CHUNK + half * 16 + kc];
      v16bf a;
      #pragma unroll
      for (int j = 0; j < 8; ++j) { a[j] = alo[j]; a[8 + j] = ahi[j]; }
      acc = __builtin_amdgcn_wmma_f32_16x16x32_bf16(false, a, false, b,
                                                    (short)0, acc, false, false);
    }
    __syncthreads();
  }
  #pragma unroll
  for (int r = 0; r < 8; ++r) {
    int m  = mt * 16 + r + half * 8;
    int nn = nt * 16 + l16;
    outF[(size_t)m * Nn + nn] = acc[r] + bias[nn];
  }
}

// ---------------------------------------------------------------------------
// batchnorm over rows (per column stats), then apply + relu
// ---------------------------------------------------------------------------
__global__ void bn_stats(const float* __restrict__ pre, float* __restrict__ st) {
  const int c = blockIdx.x, tid = threadIdx.x;
  float s = 0.f, s2 = 0.f;
  for (int n = tid; n < N_PTS; n += 256) {
    float v = pre[(size_t)n * DM + c];
    s += v; s2 += v * v;
  }
  __shared__ float sh[256], sh2[256];
  sh[tid] = s; sh2[tid] = s2; __syncthreads();
  for (int o = 128; o > 0; o >>= 1) {
    if (tid < o) { sh[tid] += sh[tid + o]; sh2[tid] += sh2[tid + o]; }
    __syncthreads();
  }
  if (tid == 0) {
    float mean = sh[0] / (float)N_PTS;
    st[c] = mean;
    st[DM + c] = sh2[0] / (float)N_PTS - mean * mean;
  }
}

__global__ void bn_relu(const float* __restrict__ pre, const float* __restrict__ st,
                        const float* __restrict__ g, const float* __restrict__ b,
                        float* __restrict__ out) {
  int e = blockIdx.x * 256 + threadIdx.x;
  if (e >= N_PTS * DM) return;
  int c = e & (DM - 1);
  float v = (pre[e] - st[c]) * rsqrtf(st[DM + c] + 1e-5f) * g[c] + b[c];
  out[e] = fmaxf(v, 0.f);
}

// ---------------------------------------------------------------------------
// positional encoding: stats of pos[:,0] (std with ddof=1), then feat + PE
// ---------------------------------------------------------------------------
__global__ void pe_stats(const float* __restrict__ pos, float* __restrict__ o2) {
  __shared__ float sh[256];
  const int tid = threadIdx.x;
  float s = 0.f;
  for (int n = tid; n < N_PTS; n += 256) s += pos[n * 3];
  sh[tid] = s; __syncthreads();
  for (int o = 128; o > 0; o >>= 1) { if (tid < o) sh[tid] += sh[tid + o]; __syncthreads(); }
  float mean = sh[0] / (float)N_PTS;
  __syncthreads();
  float s2 = 0.f;
  for (int n = tid; n < N_PTS; n += 256) { float d = pos[n * 3] - mean; s2 += d * d; }
  sh[tid] = s2; __syncthreads();
  for (int o = 128; o > 0; o >>= 1) { if (tid < o) sh[tid] += sh[tid + o]; __syncthreads(); }
  if (tid == 0) { o2[0] = mean; o2[1] = sqrtf(sh[0] / (float)(N_PTS - 1)); }
}

__global__ void add_pe(const float* __restrict__ feat, const float* __restrict__ pos,
                       const float* __restrict__ pest, bf16* __restrict__ hB) {
  int e = blockIdx.x * 256 + threadIdx.x;
  if (e >= N_PTS * DM) return;
  int n = e >> 7, d = e & 127;
  float px  = (pos[n * 3] - pest[0]) / (pest[1] + 1e-8f);
  float div = __expf((float)(d & ~1) * (-9.210340371976184f / (float)DM));
  float ang = px * div;
  float pe  = (d & 1) ? __cosf(ang) : __sinf(ang);
  hB[e] = (bf16)(feat[e] + pe);
}

// ---------------------------------------------------------------------------
// flash attention: one wave per (16-query tile, head); hd = 32 = WMMA K
// V is consumed pre-transposed (vT[dim][n]) so B-fragments load contiguously
// ---------------------------------------------------------------------------
__global__ void __launch_bounds__(32)
attn_flash(const bf16* __restrict__ qb, const bf16* __restrict__ kb,
           const bf16* __restrict__ vT, bf16* __restrict__ outB) {
  __shared__ bf16 lds_p[16 * 32];
  const int lane = threadIdx.x;
  const int half = lane >> 4, l16 = lane & 15;
  const int qt = blockIdx.x, h = blockIdx.y;
  const float scale = 0.17677669529663687f;        // 1/sqrt(32)

  // Q A-fragment (row = query l16, K = hd dims)
  const bf16* qrow = qb + (size_t)(qt * 16 + l16) * DM + h * 32 + half * 16;
  v16bf aq;
  #pragma unroll
  for (int j = 0; j < 8; ++j) { aq[j] = qrow[j]; aq[8 + j] = qrow[8 + j]; }

  const bf16* vt0 = vT + (size_t)(h * 32 + l16) * N_PTS;
  const bf16* vt1 = vT + (size_t)(h * 32 + 16 + l16) * N_PTS;

  float mrow[8], lrow[8];
  v8f o0 = {}, o1 = {};
  #pragma unroll
  for (int r = 0; r < 8; ++r) { mrow[r] = -1e30f; lrow[r] = 0.f; }

  for (int kc = 0; kc < N_PTS; kc += 32) {
    __builtin_prefetch(kb + (size_t)(kc + 32 + l16) * DM, 0, 0);
    // QK^T for two 16-key tiles (contraction over hd=32)
    v16bf bk0 = *(const v16bf*)(kb + (size_t)(kc + l16) * DM + h * 32 + half * 16);
    v16bf bk1 = *(const v16bf*)(kb + (size_t)(kc + 16 + l16) * DM + h * 32 + half * 16);
    v8f s0 = {}, s1 = {};
    s0 = __builtin_amdgcn_wmma_f32_16x16x32_bf16(false, aq, false, bk0, (short)0, s0, false, false);
    s1 = __builtin_amdgcn_wmma_f32_16x16x32_bf16(false, aq, false, bk1, (short)0, s1, false, false);

    // online softmax per query row (rows live across a 16-lane half-wave)
    #pragma unroll
    for (int r = 0; r < 8; ++r) {
      float a0 = s0[r] * scale, a1 = s1[r] * scale;
      float tmax = red16_max(fmaxf(a0, a1));
      float nm = fmaxf(mrow[r], tmax);
      float corr = __expf(mrow[r] - nm);
      float p0 = __expf(a0 - nm);
      float p1 = __expf(a1 - nm);
      float rs = red16_sum(p0 + p1);
      lrow[r] = lrow[r] * corr + rs;
      mrow[r] = nm;
      o0[r] *= corr; o1[r] *= corr;
      lds_p[(r + half * 8) * 32 + l16]      = (bf16)p0;
      lds_p[(r + half * 8) * 32 + 16 + l16] = (bf16)p1;
    }
    __syncthreads();

    // P (16x32) A-fragment from LDS, V (32x32) B-fragments (contiguous), acc O
    const bf16* prow = &lds_p[l16 * 32 + half * 16];
    v16bf pa;
    #pragma unroll
    for (int j = 0; j < 8; ++j) { pa[j] = prow[j]; pa[8 + j] = prow[8 + j]; }
    v16bf bv0 = *(const v16bf*)(vt0 + kc + half * 16);
    v16bf bv1 = *(const v16bf*)(vt1 + kc + half * 16);
    o0 = __builtin_amdgcn_wmma_f32_16x16x32_bf16(false, pa, false, bv0, (short)0, o0, false, false);
    o1 = __builtin_amdgcn_wmma_f32_16x16x32_bf16(false, pa, false, bv1, (short)0, o1, false, false);
    __syncthreads();
  }
  #pragma unroll
  for (int r = 0; r < 8; ++r) {
    float inv = 1.f / lrow[r];
    int m = qt * 16 + r + half * 8;
    outB[(size_t)m * DM + h * 32 + l16]      = (bf16)(o0[r] * inv);
    outB[(size_t)m * DM + h * 32 + 16 + l16] = (bf16)(o1[r] * inv);
  }
}

// ---------------------------------------------------------------------------
// layernorm of (a + b) per row; one wave per row
// ---------------------------------------------------------------------------
__global__ void __launch_bounds__(256)
ln_residual(const float* __restrict__ a, const float* __restrict__ b,
            const float* __restrict__ g, const float* __restrict__ be,
            float* __restrict__ outF, bf16* __restrict__ outB) {
  const int wave = threadIdx.x >> 5, lane = threadIdx.x & 31;
  const int n = blockIdx.x * 8 + wave;
  const float* ar = a + (size_t)n * DM;
  const float* br = b + (size_t)n * DM;
  float v[4]; float s = 0.f;
  #pragma unroll
  for (int i = 0; i < 4; ++i) { int c = lane + i * 32; v[i] = ar[c] + br[c]; s += v[i]; }
  s = red32_sum(s);
  float mean = s / (float)DM;
  float s2 = 0.f;
  #pragma unroll
  for (int i = 0; i < 4; ++i) { float d = v[i] - mean; s2 += d * d; }
  s2 = red32_sum(s2);
  float rstd = rsqrtf(s2 / (float)DM + 1e-5f);
  #pragma unroll
  for (int i = 0; i < 4; ++i) {
    int c = lane + i * 32;
    float o = (v[i] - mean) * rstd * g[c] + be[c];
    if (outF) outF[(size_t)n * DM + c] = o;
    if (outB) outB[(size_t)n * DM + c] = (bf16)o;
  }
}

// ---------------------------------------------------------------------------
// final sigmoid gate
// ---------------------------------------------------------------------------
__global__ void gate(const float* __restrict__ x, const float* __restrict__ y,
                     const float* __restrict__ xf, const float* __restrict__ yf,
                     float* __restrict__ out) {
  int e = blockIdx.x * 256 + threadIdx.x;
  if (e >= N_PTS * CH) return;
  float w = 1.f / (1.f + __expf(-(xf[e] + yf[e])));
  out[e] = 2.f * x[e] * w + 2.f * y[e] * (1.f - w);
}

// ---------------------------------------------------------------------------
// host orchestration
// ---------------------------------------------------------------------------
extern "C" void kernel_launch(void* const* d_in, const int* in_sizes, int n_in,
                              void* d_out, int out_size, void* d_ws, size_t ws_size,
                              hipStream_t stream) {
  (void)in_sizes; (void)n_in; (void)out_size; (void)ws_size;
  const float* x    = (const float*)d_in[0];
  const float* y    = (const float*)d_in[1];
  const float* pos  = (const float*)d_in[2];
  const int*   idx  = (const int*)d_in[3];
  const unsigned char* mask = (const unsigned char*)d_in[4];
  const float* Wconv[2] = {(const float*)d_in[5], (const float*)d_in[9]};
  const float* bconv[2] = {(const float*)d_in[6], (const float*)d_in[10]};
  const float* bng[2]   = {(const float*)d_in[7], (const float*)d_in[11]};
  const float* bnb[2]   = {(const float*)d_in[8], (const float*)d_in[12]};
  const float* Wq = (const float*)d_in[13]; const float* bq = (const float*)d_in[14];
  const float* Wk = (const float*)d_in[15]; const float* bk = (const float*)d_in[16];
  const float* Wv = (const float*)d_in[17]; const float* bv = (const float*)d_in[18];
  const float* Wo = (const float*)d_in[19]; const float* bo = (const float*)d_in[20];
  const float* ln1g = (const float*)d_in[21]; const float* ln1b = (const float*)d_in[22];
  const float* W1 = (const float*)d_in[23]; const float* b1 = (const float*)d_in[24];
  const float* W2 = (const float*)d_in[25]; const float* b2 = (const float*)d_in[26];
  const float* ln2g = (const float*)d_in[27]; const float* ln2b = (const float*)d_in[28];
  const float* featIn[2] = {x, y};

  char* base = (char*)d_ws;
  size_t off = 0;
  auto alloc = [&](size_t bytes) -> char* {
    char* p = base + off;
    off += (bytes + 255) & ~(size_t)255;
    return p;
  };

  bf16* Abuf   = (bf16*)alloc((size_t)N_PTS * NCELL * CH * 2); // 64 MB
  bf16* WfT    = (bf16*)alloc((size_t)NCELL * CH * DM * 2);
  bf16* WqB    = (bf16*)alloc(DM * DM * 2);
  bf16* WkB    = (bf16*)alloc(DM * DM * 2);
  bf16* WvB    = (bf16*)alloc(DM * DM * 2);
  bf16* WoB    = (bf16*)alloc(DM * DM * 2);
  bf16* W1B    = (bf16*)alloc(FFN * DM * 2);
  bf16* W2B    = (bf16*)alloc(DM * FFN * 2);
  float* pre    = (float*)alloc((size_t)N_PTS * DM * 4);
  float* stats  = (float*)alloc(2 * DM * 4);
  float* pest   = (float*)alloc(256);
  float* feat   = (float*)alloc((size_t)N_PTS * DM * 4);
  bf16*  hB     = (bf16*)alloc((size_t)N_PTS * DM * 2);
  bf16*  qB     = (bf16*)alloc((size_t)N_PTS * DM * 2);
  bf16*  kB     = (bf16*)alloc((size_t)N_PTS * DM * 2);
  bf16*  vTB    = (bf16*)alloc((size_t)DM * N_PTS * 2);        // V transposed
  bf16*  attnB  = (bf16*)alloc((size_t)N_PTS * DM * 2);
  float* oproj  = (float*)alloc((size_t)N_PTS * DM * 4);
  float* feat2  = (float*)alloc((size_t)N_PTS * DM * 4);
  bf16*  feat2B = (bf16*)alloc((size_t)N_PTS * DM * 2);
  bf16*  midB   = (bf16*)alloc((size_t)N_PTS * FFN * 2);
  float* ffout  = (float*)alloc((size_t)N_PTS * DM * 4);
  float* sfOut[2];
  sfOut[0] = (float*)alloc((size_t)N_PTS * DM * 4);
  sfOut[1] = (float*)alloc((size_t)N_PTS * DM * 4);

  const int EW = N_PTS * DM;                 // 524288 elementwise grid
  dim3 b256(256);

  // one-time weight conversions + PE stats
  convert_f32_bf16<<<(DM * DM + 255) / 256, b256, 0, stream>>>(Wq, WqB, DM * DM);
  convert_f32_bf16<<<(DM * DM + 255) / 256, b256, 0, stream>>>(Wk, WkB, DM * DM);
  convert_f32_bf16<<<(DM * DM + 255) / 256, b256, 0, stream>>>(Wv, WvB, DM * DM);
  convert_f32_bf16<<<(DM * DM + 255) / 256, b256, 0, stream>>>(Wo, WoB, DM * DM);
  convert_f32_bf16<<<(FFN * DM + 255) / 256, b256, 0, stream>>>(W1, W1B, FFN * DM);
  convert_f32_bf16<<<(DM * FFN + 255) / 256, b256, 0, stream>>>(W2, W2B, DM * FFN);
  pe_stats<<<1, b256, 0, stream>>>(pos, pest);

  for (int s = 0; s < 2; ++s) {
    // --- continuous conv (TDM-staged GEMM) ---
    convert_Wf_T<<<(NCELL * CH * DM + 255) / 256, b256, 0, stream>>>(Wconv[s], WfT);
    cconv_gather<<<N_PTS, b256, 0, stream>>>(featIn[s], pos, idx, mask, Abuf);
    gemm_cconv_tdm<<<dim3(N_PTS / 64, DM / 16), dim3(128), 0, stream>>>(
        Abuf, WfT, bconv[s], pre);
    bn_stats<<<DM, b256, 0, stream>>>(pre, stats);
    bn_relu<<<(EW + 255) / 256, b256, 0, stream>>>(pre, stats, bng[s], bnb[s], feat);

    // --- transformer block ---
    add_pe<<<(EW + 255) / 256, b256, 0, stream>>>(feat, pos, pest, hB);
    gemm_bf16<<<dim3(N_PTS / 64, DM / 16), dim3(128), 0, stream>>>(
        hB, WqB, bq, nullptr, nullptr, qB, N_PTS, DM, DM, 0, 0);
    gemm_bf16<<<dim3(N_PTS / 64, DM / 16), dim3(128), 0, stream>>>(
        hB, WkB, bk, nullptr, nullptr, kB, N_PTS, DM, DM, 0, 0);
    gemm_bf16<<<dim3(N_PTS / 64, DM / 16), dim3(128), 0, stream>>>(
        hB, WvB, bv, nullptr, nullptr, vTB, N_PTS, DM, DM, 0, 1);
    attn_flash<<<dim3(N_PTS / 16, 4), dim3(32), 0, stream>>>(qB, kB, vTB, attnB);
    gemm_bf16<<<dim3(N_PTS / 64, DM / 16), dim3(128), 0, stream>>>(
        attnB, WoB, bo, nullptr, oproj, nullptr, N_PTS, DM, DM, 0, 0);
    ln_residual<<<N_PTS / 8, b256, 0, stream>>>(feat, oproj, ln1g, ln1b, feat2, feat2B);
    gemm_bf16<<<dim3(N_PTS / 64, FFN / 16), dim3(128), 0, stream>>>(
        feat2B, W1B, b1, nullptr, nullptr, midB, N_PTS, DM, FFN, 1, 0);
    gemm_bf16<<<dim3(N_PTS / 64, DM / 16), dim3(128), 0, stream>>>(
        midB, W2B, b2, nullptr, ffout, nullptr, N_PTS, FFN, DM, 0, 0);
    ln_residual<<<N_PTS / 8, b256, 0, stream>>>(feat2, ffout, ln2g, ln2b, sfOut[s], nullptr);
  }

  gate<<<(EW + 255) / 256, b256, 0, stream>>>(x, y, sfOut[0], sfOut[1], (float*)d_out);
}
